// Encoder_14474039787789
// MI455X (gfx1250) — matrix operand order repaired and checked by reading, not compiled, not run on hardware
//
#include <hip/hip_runtime.h>
#include <hip/hip_bf16.h>

// ---------------- problem constants ----------------
#define NLAYER 12
#define DMODEL 768
#define NHEAD  12
#define FDIM   3072
#define BATCH  8
#define SEQ    784
#define HDIM   64
#define NTOK   (BATCH * SEQ)   // 6272 = 49 * 128

typedef __attribute__((ext_vector_type(16))) __bf16 v16bf;
typedef __attribute__((ext_vector_type(8)))  float  v8f;
typedef __attribute__((ext_vector_type(4)))  unsigned int v4u;
typedef __attribute__((ext_vector_type(8)))  int v8i;
typedef __attribute__((ext_vector_type(4)))  int v4i;

#if defined(__gfx1250__) && __has_builtin(__builtin_amdgcn_tensor_load_to_lds)
#define HAVE_TDM 1
#else
#define HAVE_TDM 0
#endif

union AF { v16bf v; uint4 q[2]; };

// Load a 16x32 bf16 A-fragment from row-major storage (tile at base, row stride ld elems).
// ISA 7.12.2: lanes 0-15 hold M=lane, K 0..7 & 16..23; lanes 16-31 hold K 8..15 & 24..31.
__device__ inline v16bf load_frag_rowmajor(const __bf16* base, int ld) {
    int lane = threadIdx.x & 31;
    int m = lane & 15, h = lane >> 4;
    const __bf16* p = base + m * ld;
    AF f;
    f.q[0] = *(const uint4*)(p + 8 * h);
    f.q[1] = *(const uint4*)(p + 16 + 8 * h);
    return f.v;
}

// Load a 32x16 bf16 B-fragment from TRANSPOSED storage baseT[n][k] (row stride ld elems).
__device__ inline v16bf load_frag_T(const __bf16* baseT, int ld) {
    int lane = threadIdx.x & 31;
    int n = lane & 15, h = lane >> 4;
    const __bf16* p = baseT + n * ld;
    AF f;
    f.q[0] = *(const uint4*)(p + 8 * h);
    f.q[1] = *(const uint4*)(p + 16 + 8 * h);
    return f.v;
}

__device__ inline v8f wmma_bf16(v16bf a, v16bf b, v8f c) {
    return __builtin_amdgcn_wmma_f32_16x16x32_bf16(false, a, false, b, (short)0, c, false, false);
}

// Async global->LDS 16-byte copy (GLOBAL_LOAD_ASYNC_TO_LDS_B128, ASYNCcnt-tracked).
// LDS dest address = low 32 bits of the flat shared-memory pointer (ISA 10.2 aperture rule).
__device__ inline void async_copy_b128(const void* gsrc, void* lds_dst) {
    unsigned lo = (unsigned)(unsigned long long)(uintptr_t)lds_dst;
    unsigned long long ga = (unsigned long long)(uintptr_t)gsrc;
    asm volatile("global_load_async_to_lds_b128 %0, %1, off"
                 :: "v"(lo), "v"(ga) : "memory");
}
__device__ inline void async_wait0() {
    asm volatile("s_wait_asynccnt 0" ::: "memory");
}

#if HAVE_TDM
// TDM: DMA one 32(k) x 128(row) bf16 tile (row stride = rowstride elems) into LDS,
// hardware-padding each 64B row by 16B -> LDS row stride 40 bf16 (matches fragment loads).
// D# packing per CDNA5 ISA ch.8 (group0: count/lds/global/type, group1: dims+pad).
__device__ inline void tdm_load_tile(const __bf16* gsrc, void* lds_dst, int rowstride) {
    unsigned lds = (unsigned)(unsigned long long)(uintptr_t)lds_dst;
    unsigned long long ga = (unsigned long long)(uintptr_t)gsrc;
    v4u g0;
    g0.x = 1u;                                            // count=1, user descriptor
    g0.y = lds;                                           // lds_addr (bytes)
    g0.z = (unsigned)ga;                                  // global_addr[31:0]
    g0.w = (unsigned)((ga >> 32) & 0x01FFFFFFu) | (2u << 30);  // addr[56:32] | type=2
    unsigned td0 = (unsigned)rowstride;                   // tensor_dim0
    unsigned td1 = 128u;                                  // tensor_dim1 (tile fits exactly)
    v8i g1;
    g1[0] = (int)((1u << 16)        // data_size = 1 -> 2 bytes
                | (1u << 20)        // pad_enable
                | (3u << 22)        // pad_interval: every 16 DWORDs (64B)
                | (3u << 25));      // pad_amount: 4 DWORDs (16B)
    g1[1] = (int)((td0 & 0xFFFFu) << 16);                               // tensor_dim0 lo16
    g1[2] = (int)(((td0 >> 16) & 0xFFFFu) | ((td1 & 0xFFFFu) << 16));   // dim0 hi | dim1 lo
    g1[3] = (int)(((td1 >> 16) & 0xFFFFu) | (32u << 16));               // dim1 hi | tile_dim0=32
    g1[4] = (int)128u;                                    // tile_dim1=128, tile_dim2=0
    g1[5] = (int)td0;                                     // tensor_dim0_stride lo32
    g1[6] = 0;                                            // stride hi16 | dim1_stride lo16
    g1[7] = 0;
    v4i zz = {0, 0, 0, 0};
#if __clang_major__ >= 23
    v8i zz8 = {0, 0, 0, 0, 0, 0, 0, 0};
    __builtin_amdgcn_tensor_load_to_lds(g0, g1, zz, zz, zz8, 0);
#else
    __builtin_amdgcn_tensor_load_to_lds(g0, g1, zz, zz, 0);
#endif
}
#endif

// ---------------- LayerNorm (row-wise over D=768), 256 thr / row ----------------
template <bool OUT_BF16>
__global__ __launch_bounds__(256) void ln_kernel(const float* __restrict__ x,
                                                 const float* __restrict__ s,
                                                 const float* __restrict__ b,
                                                 void* __restrict__ out) {
    int row = blockIdx.x;
    int t = threadIdx.x;
    const float* xr = x + (size_t)row * DMODEL;
    float v0 = xr[t], v1 = xr[t + 256], v2 = xr[t + 512];

    __shared__ float red[8];
    __shared__ float mv[2];

    float sum = v0 + v1 + v2;
    #pragma unroll
    for (int off = 16; off > 0; off >>= 1) sum += __shfl_down(sum, off, 32);
    if ((t & 31) == 0) red[t >> 5] = sum;
    __syncthreads();
    if (t == 0) {
        float tt = 0.f;
        #pragma unroll
        for (int i = 0; i < 8; i++) tt += red[i];
        mv[0] = tt * (1.0f / DMODEL);
    }
    __syncthreads();
    float mean = mv[0];
    float d0 = v0 - mean, d1 = v1 - mean, d2 = v2 - mean;
    float sq = d0 * d0 + d1 * d1 + d2 * d2;
    #pragma unroll
    for (int off = 16; off > 0; off >>= 1) sq += __shfl_down(sq, off, 32);
    if ((t & 31) == 0) red[t >> 5] = sq;
    __syncthreads();
    if (t == 0) {
        float tt = 0.f;
        #pragma unroll
        for (int i = 0; i < 8; i++) tt += red[i];
        mv[1] = rsqrtf(tt * (1.0f / DMODEL) + 1e-6f);
    }
    __syncthreads();
    float rs = mv[1];

    float y0 = d0 * rs * s[t] + b[t];
    float y1 = d1 * rs * s[t + 256] + b[t + 256];
    float y2 = d2 * rs * s[t + 512] + b[t + 512];
    if (OUT_BF16) {
        __bf16* o = (__bf16*)out + (size_t)row * DMODEL;
        o[t] = (__bf16)y0; o[t + 256] = (__bf16)y1; o[t + 512] = (__bf16)y2;
    } else {
        float* o = (float*)out + (size_t)row * DMODEL;
        o[t] = y0; o[t + 256] = y1; o[t + 512] = y2;
    }
}

// ---------------- weight transpose + fp32->bf16: W[K][N] -> Wt[N][K] ----------------
__global__ __launch_bounds__(256) void wconv_kernel(const float* __restrict__ W,
                                                    __bf16* __restrict__ Wt,
                                                    int K, int N) {
    __shared__ float tile[32][33];
    int kb = blockIdx.x * 32, nb = blockIdx.y * 32;
    int c = threadIdx.x & 31, r = threadIdx.x >> 5;
    #pragma unroll
    for (int i = 0; i < 4; i++)
        tile[r + 8 * i][c] = W[(size_t)(kb + r + 8 * i) * N + nb + c];
    __syncthreads();
    #pragma unroll
    for (int i = 0; i < 4; i++)
        Wt[(size_t)(nb + r + 8 * i) * K + kb + c] = (__bf16)tile[c][r + 8 * i];
}

// ---------------- bf16 WMMA GEMM: C = A(MxK bf16) * Wt(NxK bf16, pre-transposed) + bias --------
// Double-buffered LDS, staged by TDM (tensor_load_to_lds) or async b128 loads; loads for
// tile t+1 overlap the WMMAs of tile t; one barrier per k-step.
// EPI 0: bf16 (acc+bias)*scale | EPI 1: bf16 exact-erf GELU | EPI 2: f32 acc+bias+resid
template <int EPI>
__global__ __launch_bounds__(256) void gemm_bf16_kernel(const __bf16* __restrict__ A,
                                                        const __bf16* __restrict__ BT,
                                                        const float* __restrict__ bias,
                                                        const float* __restrict__ resid,
                                                        void* __restrict__ Cout,
                                                        int M, int N, int K, float scale) {
    __shared__ __bf16 As[2][128 * 40];   // [m][k], padded stride 40
    __shared__ __bf16 Bt[2][128 * 40];   // [n][k], padded stride 40

    int tid = threadIdx.x;
    int wave = tid >> 5;
    int wrow = wave >> 2, wcol = wave & 3;           // 2 x 4 wave grid
    int bm = blockIdx.x * 128, bn = blockIdx.y * 128;

    v8f acc[4][2];
    #pragma unroll
    for (int i = 0; i < 4; i++)
        #pragma unroll
        for (int j = 0; j < 2; j++)
            #pragma unroll
            for (int r = 0; r < 8; r++) acc[i][j][r] = 0.f;

    int srow = tid >> 1, scb = (tid & 1) * 16;       // 128 rows x 32 k, 16-elem chunks
    const __bf16* ag = A  + (size_t)(bm + srow) * K + scb;
    const __bf16* bg = BT + (size_t)(bn + srow) * K + scb;

    auto stage = [&](int kt, int buf) {
#if HAVE_TDM
        if (wave == 0) {
            tdm_load_tile(A  + (size_t)bm * K + kt, &As[buf][0], K);
            tdm_load_tile(BT + (size_t)bn * K + kt, &Bt[buf][0], K);
        }
#else
        async_copy_b128(ag + kt,     &As[buf][srow * 40 + scb]);
        async_copy_b128(ag + kt + 8, &As[buf][srow * 40 + scb + 8]);
        async_copy_b128(bg + kt,     &Bt[buf][srow * 40 + scb]);
        async_copy_b128(bg + kt + 8, &Bt[buf][srow * 40 + scb + 8]);
#endif
    };
    auto stage_wait = [&]() {
#if HAVE_TDM
        if (wave == 0) __builtin_amdgcn_s_wait_tensorcnt((short)0);
#else
        async_wait0();
#endif
    };

    stage(0, 0);
    stage_wait();
    __syncthreads();

    int nk = K >> 5;
    for (int ik = 0; ik < nk; ik++) {
        int buf = ik & 1;
        if (ik + 1 < nk) stage((ik + 1) << 5, buf ^ 1);    // overlap next tile with compute
        if (ik + 2 < nk) {                                  // keep L2 ahead of the DMA engine
            __builtin_prefetch(ag + ((ik + 2) << 5), 0, 1);
            __builtin_prefetch(bg + ((ik + 2) << 5), 0, 1);
        }

        v16bf af[4], bfr[2];
        #pragma unroll
        for (int mt = 0; mt < 4; mt++)
            af[mt] = load_frag_rowmajor(&As[buf][(wrow * 64 + mt * 16) * 40], 40);
        #pragma unroll
        for (int nt = 0; nt < 2; nt++)
            bfr[nt] = load_frag_T(&Bt[buf][(wcol * 32 + nt * 16) * 40], 40);
        #pragma unroll
        for (int mt = 0; mt < 4; mt++)
            #pragma unroll
            for (int nt = 0; nt < 2; nt++)
                acc[mt][nt] = wmma_bf16(af[mt], bfr[nt], acc[mt][nt]);

        stage_wait();
        __syncthreads();
    }

    int lane = tid & 31;
    int n = lane & 15, h = lane >> 4;
    #pragma unroll
    for (int mt = 0; mt < 4; mt++)
        #pragma unroll
        for (int nt = 0; nt < 2; nt++) {
            int col = bn + wcol * 32 + nt * 16 + n;
            float bv = bias[col];
            #pragma unroll
            for (int r = 0; r < 8; r++) {
                int row = bm + wrow * 64 + mt * 16 + r + 8 * h;
                float c = acc[mt][nt][r] + bv;
                if (EPI == 0) {
                    ((__bf16*)Cout)[(size_t)row * N + col] = (__bf16)(c * scale);
                } else if (EPI == 1) {
                    float g = 0.5f * c * (1.0f + erff(c * 0.70710678118654752f));
                    ((__bf16*)Cout)[(size_t)row * N + col] = (__bf16)g;
                } else {
                    ((float*)Cout)[(size_t)row * N + col] = c + resid[(size_t)row * N + col];
                }
            }
        }
}

// ---------------- flash attention: WG = (qblk of 64 rows, head, batch), 4 waves ----------------
__global__ __launch_bounds__(128) void attn_kernel(const __bf16* __restrict__ q,
                                                   const __bf16* __restrict__ k,
                                                   const __bf16* __restrict__ v,
                                                   __bf16* __restrict__ ctx) {
    constexpr int KT = 72;   // Kt row stride ([key][feat])
    constexpr int VT = 40;   // Vt row stride ([feat][key], transposed)
    __shared__ __bf16 Kt[32 * KT];
    __shared__ __bf16 Vt[64 * VT];
    __shared__ float  sS[4][16 * 33];
    __shared__ __bf16 pS[4][16 * 40];
    __shared__ float  rowf[4][16];
    __shared__ float  rowl[4][16];

    int b = blockIdx.z, hh = blockIdx.y, qblk = blockIdx.x;
    int tid = threadIdx.x;
    int wave = tid >> 5, lane = tid & 31;
    int m = lane & 15, h = lane >> 4;

    int qt0 = qblk * 64 + wave * 16;

    v16bf qf[2];
    {
        int tok = qt0 + m; if (tok >= SEQ) tok = SEQ - 1;
        const __bf16* qp = q + (size_t)(b * SEQ + tok) * DMODEL + hh * HDIM;
        AF f0, f1;
        f0.q[0] = *(const uint4*)(qp + 8 * h);
        f0.q[1] = *(const uint4*)(qp + 16 + 8 * h);
        f1.q[0] = *(const uint4*)(qp + 32 + 8 * h);
        f1.q[1] = *(const uint4*)(qp + 48 + 8 * h);
        qf[0] = f0.v; qf[1] = f1.v;
    }

    v8f o[4];
    #pragma unroll
    for (int j = 0; j < 4; j++)
        #pragma unroll
        for (int r = 0; r < 8; r++) o[j][r] = 0.f;
    float mrun = -1e30f, lrun = 0.f;

    int skey = tid >> 2;
    int sfb  = (tid & 3) * 16;

    for (int kb = 0; kb < 25; kb++) {      // 25 * 32 = 800 >= 784
        int key0 = kb * 32;
        {
            int tok = key0 + skey;
            if (tok < SEQ) {
                const __bf16* kp = k + (size_t)(b * SEQ + tok) * DMODEL + hh * HDIM + sfb;
                uint4 k0 = *(const uint4*)kp;
                uint4 k1 = *(const uint4*)(kp + 8);
                *(uint4*)&Kt[skey * KT + sfb]     = k0;
                *(uint4*)&Kt[skey * KT + sfb + 8] = k1;
                const __bf16* vp = v + (size_t)(b * SEQ + tok) * DMODEL + hh * HDIM + sfb;
                #pragma unroll
                for (int i = 0; i < 16; i++) Vt[(sfb + i) * VT + skey] = vp[i];
                if (tok + 32 < SEQ) {
                    __builtin_prefetch(kp + 32 * DMODEL, 0, 1);
                    __builtin_prefetch(vp + 32 * DMODEL, 0, 1);
                }
            } else {
                uint4 z = make_uint4(0, 0, 0, 0);
                *(uint4*)&Kt[skey * KT + sfb]     = z;
                *(uint4*)&Kt[skey * KT + sfb + 8] = z;
                #pragma unroll
                for (int i = 0; i < 16; i++) Vt[(sfb + i) * VT + skey] = (__bf16)0.f;
            }
        }
        __syncthreads();

        v8f c0, c1;
        #pragma unroll
        for (int r = 0; r < 8; r++) { c0[r] = 0.f; c1[r] = 0.f; }
        v16bf b00 = load_frag_T(&Kt[0 * KT + 0],  KT);
        v16bf b01 = load_frag_T(&Kt[0 * KT + 32], KT);
        v16bf b10 = load_frag_T(&Kt[16 * KT + 0],  KT);
        v16bf b11 = load_frag_T(&Kt[16 * KT + 32], KT);
        c0 = wmma_bf16(qf[0], b00, c0); c0 = wmma_bf16(qf[1], b01, c0);
        c1 = wmma_bf16(qf[0], b10, c1); c1 = wmma_bf16(qf[1], b11, c1);

        #pragma unroll
        for (int r = 0; r < 8; r++) {
            sS[wave][(r + 8 * h) * 33 + m]      = c0[r];
            sS[wave][(r + 8 * h) * 33 + m + 16] = c1[r];
        }

        {
            int row = m;
            int ks = h * 16;
            float mx = -1e30f;
            float pv[16];
            #pragma unroll
            for (int j = 0; j < 16; j++) {
                int gk = key0 + ks + j;
                float sv = (gk < SEQ) ? sS[wave][row * 33 + ks + j] : -1e30f;
                pv[j] = sv;
                mx = fmaxf(mx, sv);
            }
            mx = fmaxf(mx, __shfl_xor(mx, 16, 32));
            float mnew = fmaxf(mrun, mx);
            float f = __expf(mrun - mnew);
            float sum = 0.f;
            #pragma unroll
            for (int j = 0; j < 16; j++) {
                float p = (pv[j] > -1e29f) ? __expf(pv[j] - mnew) : 0.f;
                sum += p;
                pS[wave][row * 40 + ks + j] = (__bf16)p;
            }
            sum += __shfl_xor(sum, 16, 32);
            lrun = lrun * f + sum;
            mrun = mnew;
            rowf[wave][row] = f;
        }

        float fr[8];
        #pragma unroll
        for (int r = 0; r < 8; r++) fr[r] = rowf[wave][r + 8 * h];
        #pragma unroll
        for (int j = 0; j < 4; j++)
            #pragma unroll
            for (int r = 0; r < 8; r++) o[j][r] *= fr[r];

        v16bf pf = load_frag_rowmajor(&pS[wave][0], 40);
        #pragma unroll
        for (int j = 0; j < 4; j++) {
            v16bf vf = load_frag_T(&Vt[(j * 16) * VT], VT);
            o[j] = wmma_bf16(pf, vf, o[j]);
        }
        __syncthreads();
    }

    rowl[wave][m] = lrun;
    float inv[8];
    #pragma unroll
    for (int r = 0; r < 8; r++) inv[r] = 1.0f / rowl[wave][r + 8 * h];

    int n = m;
    #pragma unroll
    for (int j = 0; j < 4; j++)
        #pragma unroll
        for (int r = 0; r < 8; r++) {
            int tok = qt0 + r + 8 * h;
            if (tok < SEQ)
                ctx[(size_t)(b * SEQ + tok) * DMODEL + hh * HDIM + j * 16 + n] =
                    (__bf16)(o[j][r] * inv[r]);
        }
}

// ---------------- host orchestration ----------------
extern "C" void kernel_launch(void* const* d_in, const int* in_sizes, int n_in,
                              void* d_out, int out_size, void* d_ws, size_t ws_size,
                              hipStream_t stream) {
    const float* x     = (const float*)d_in[0];
    const float* ln1_s = (const float*)d_in[1];
    const float* ln1_b = (const float*)d_in[2];
    const float* wq    = (const float*)d_in[3];
    const float* bq    = (const float*)d_in[4];
    const float* wk    = (const float*)d_in[5];
    const float* bk    = (const float*)d_in[6];
    const float* wv    = (const float*)d_in[7];
    const float* bv    = (const float*)d_in[8];
    const float* wo    = (const float*)d_in[9];
    const float* bo    = (const float*)d_in[10];
    const float* ln2_s = (const float*)d_in[11];
    const float* ln2_b = (const float*)d_in[12];
    const float* w1    = (const float*)d_in[13];
    const float* b1    = (const float*)d_in[14];
    const float* w2    = (const float*)d_in[15];
    const float* b2    = (const float*)d_in[16];
    const float* lnf_s = (const float*)d_in[17];
    const float* lnf_b = (const float*)d_in[18];

    float* hres = (float*)d_out;                       // residual stream lives in d_out

    const size_t TD = (size_t)NTOK * DMODEL;
    __bf16* y   = (__bf16*)d_ws;
    __bf16* qb  = y   + TD;
    __bf16* kb  = qb  + TD;
    __bf16* vb  = kb  + TD;
    __bf16* cb  = vb  + TD;
    __bf16* gb  = cb  + TD;                            // NTOK x FDIM
    __bf16* wt  = gb  + (size_t)NTOK * FDIM;           // transposed bf16 weight (max F*D)

    hipMemcpyAsync(hres, x, TD * sizeof(float), hipMemcpyDeviceToDevice, stream);

    dim3 gD(NTOK / 128, DMODEL / 128);                 // 49 x 6
    dim3 gF(NTOK / 128, FDIM / 128);                   // 49 x 24
    dim3 gA(13, NHEAD, BATCH);                         // ceil(784/64)=13 q-blocks
    dim3 tDD(DMODEL / 32, DMODEL / 32);
    dim3 tDF(DMODEL / 32, FDIM / 32);
    dim3 tFD(FDIM / 32, DMODEL / 32);

    for (int l = 0; l < NLAYER; l++) {
        const float* Wq = wq + (size_t)l * DMODEL * DMODEL;
        const float* Wk = wk + (size_t)l * DMODEL * DMODEL;
        const float* Wv = wv + (size_t)l * DMODEL * DMODEL;
        const float* Wo = wo + (size_t)l * DMODEL * DMODEL;
        const float* W1 = w1 + (size_t)l * DMODEL * FDIM;
        const float* W2 = w2 + (size_t)l * FDIM * DMODEL;
        const float* Bq = bq + (size_t)l * DMODEL;
        const float* Bk = bk + (size_t)l * DMODEL;
        const float* Bv = bv + (size_t)l * DMODEL;
        const float* Bo = bo + (size_t)l * DMODEL;
        const float* B1 = b1 + (size_t)l * FDIM;
        const float* B2 = b2 + (size_t)l * DMODEL;

        ln_kernel<true><<<NTOK, 256, 0, stream>>>(hres, ln1_s + (size_t)l * DMODEL,
                                                  ln1_b + (size_t)l * DMODEL, y);
        wconv_kernel<<<tDD, 256, 0, stream>>>(Wq, wt, DMODEL, DMODEL);
        gemm_bf16_kernel<0><<<gD, 256, 0, stream>>>(y, wt, Bq, nullptr, qb,
                                                    NTOK, DMODEL, DMODEL, 0.125f);
        wconv_kernel<<<tDD, 256, 0, stream>>>(Wk, wt, DMODEL, DMODEL);
        gemm_bf16_kernel<0><<<gD, 256, 0, stream>>>(y, wt, Bk, nullptr, kb,
                                                    NTOK, DMODEL, DMODEL, 1.0f);
        wconv_kernel<<<tDD, 256, 0, stream>>>(Wv, wt, DMODEL, DMODEL);
        gemm_bf16_kernel<0><<<gD, 256, 0, stream>>>(y, wt, Bv, nullptr, vb,
                                                    NTOK, DMODEL, DMODEL, 1.0f);
        attn_kernel<<<gA, 128, 0, stream>>>(qb, kb, vb, cb);
        wconv_kernel<<<tDD, 256, 0, stream>>>(Wo, wt, DMODEL, DMODEL);
        gemm_bf16_kernel<2><<<gD, 256, 0, stream>>>(cb, wt, Bo, hres, hres,
                                                    NTOK, DMODEL, DMODEL, 1.0f);
        ln_kernel<true><<<NTOK, 256, 0, stream>>>(hres, ln2_s + (size_t)l * DMODEL,
                                                  ln2_b + (size_t)l * DMODEL, y);
        wconv_kernel<<<tDF, 256, 0, stream>>>(W1, wt, DMODEL, FDIM);
        gemm_bf16_kernel<1><<<gF, 256, 0, stream>>>(y, wt, B1, nullptr, gb,
                                                    NTOK, FDIM, DMODEL, 1.0f);
        wconv_kernel<<<tFD, 256, 0, stream>>>(W2, wt, FDIM, DMODEL);
        gemm_bf16_kernel<2><<<gD, 256, 0, stream>>>(gb, wt, B2, hres, hres,
                                                    NTOK, DMODEL, FDIM, 1.0f);
    }
    ln_kernel<false><<<NTOK, 256, 0, stream>>>(hres, lnf_s, lnf_b, hres);
}
